// BigramLanguageModel_17712445129027
// MI455X (gfx1250) — compile-verified, matrix operand order checked by reading, not run on hardware
//
#include <hip/hip_runtime.h>
#include <math.h>

// ---------------- types ----------------
typedef __attribute__((ext_vector_type(16))) __bf16  v16bf;
typedef __attribute__((ext_vector_type(8)))  float   v8f;
typedef __attribute__((ext_vector_type(4)))  float   v4f;
typedef __attribute__((ext_vector_type(4)))  unsigned int v4u;
typedef int v4i_ __attribute__((vector_size(16)));   // matches builtin's V4i

union Frag { v16bf v; v4u u[2]; };

// gfx1250 async global->LDS copy (ASYNCcnt-tracked). Probed; sync fallback.
#if defined(__gfx1250__) && __has_builtin(__builtin_amdgcn_global_load_async_to_lds_b128)
#define HAS_ASYNC 1
typedef __attribute__((address_space(1))) v4i_* as1_v4i_p;
typedef __attribute__((address_space(3))) v4i_* as3_v4i_p;
#else
#define HAS_ASYNC 0
#endif

__device__ __forceinline__ unsigned short f2bf(float f) {
  unsigned int u = __float_as_uint(f);
  unsigned int r = u + 0x7FFFu + ((u >> 16) & 1u);   // round-to-nearest-even
  return (unsigned short)(r >> 16);
}

__device__ __forceinline__ void cp16(const void* g, void* l) {
#if HAS_ASYNC
  __builtin_amdgcn_global_load_async_to_lds_b128((as1_v4i_p)g, (as3_v4i_p)l, 0, 0);
#else
  *(v4u*)l = *(const v4u*)g;
#endif
}

__device__ __forceinline__ void wait_async_8() {
#if HAS_ASYNC
#if __has_builtin(__builtin_amdgcn_s_wait_asynccnt)
  __builtin_amdgcn_s_wait_asynccnt(8);
#else
  asm volatile("s_wait_asynccnt 0x8" ::: "memory");
#endif
#endif
}
__device__ __forceinline__ void wait_async_4() {
#if HAS_ASYNC
#if __has_builtin(__builtin_amdgcn_s_wait_asynccnt)
  __builtin_amdgcn_s_wait_asynccnt(4);
#else
  asm volatile("s_wait_asynccnt 0x4" ::: "memory");
#endif
#endif
}
__device__ __forceinline__ void wait_async_0() {
#if HAS_ASYNC
#if __has_builtin(__builtin_amdgcn_s_wait_asynccnt)
  __builtin_amdgcn_s_wait_asynccnt(0);
#else
  asm volatile("s_wait_asynccnt 0x0" ::: "memory");
#endif
#endif
}

// ---------------- WMMA GEMM ----------------
// C[M,N] = (residual?) + A_bf16[M,K] @ B + bias, opt ReLU, out fp32 or bf16.
// BMODE 1: B = Wt bf16 [N,K] (pre-transposed) -> raw async tile copies, dbl-buffered LDS.
// BMODE 0: B = Wf fp32 [K,N] (fallback)       -> in-kernel convert + LDS transpose.
// Block tile 128x128, K-step 32; 256 thr = 8 waves; wave tile 64x32 = 4x2 wmma.
#define BM 128
#define BN 128
#define BK 32
#define TSTR 40   // ushort LDS row stride (80 B: 16B-aligned, conflict-padded)

template <int BMODE>
__global__ __launch_bounds__(256) void gemm_wmma(
    const unsigned short* __restrict__ Abf,
    const unsigned short* __restrict__ Wt,
    const float* __restrict__ Wf,
    const float* __restrict__ bias,
    const float* __restrict__ residual,
    float* __restrict__ C,
    unsigned short* __restrict__ Cbf,
    int M, int N, int K, int relu)
{
  __shared__ unsigned short As[(BMODE ? 2 : 1) * BM * TSTR];
  __shared__ unsigned short Bs[(BMODE ? 2 : 1) * BM * TSTR];

  const int tid  = threadIdx.x;
  const int lane = tid & 31;
  const int wid  = tid >> 5;
  const int wm   = wid >> 2;      // 0..1  (64-row strip)
  const int wn   = wid & 3;       // 0..3  (32-col strip)
  const int h    = lane >> 4;     // lane half
  const int l15  = lane & 15;

  const int m0 = blockIdx.y * BM;
  const int n0 = blockIdx.x * BN;

  v8f acc[4][2];
  #pragma unroll
  for (int mi = 0; mi < 4; ++mi)
    #pragma unroll
    for (int ni = 0; ni < 2; ++ni)
      #pragma unroll
      for (int e = 0; e < 8; ++e) acc[mi][ni][e] = 0.0f;

  const int nk = K / BK;

  // fragments per ISA 16-bit A layout: lane half h holds K = h*8..h*8+7, 16+h*8..+7
  auto compute = [&](const unsigned short* Asb, const unsigned short* Bsb) {
    Frag a[4], b[2];
    #pragma unroll
    for (int mi = 0; mi < 4; ++mi) {
      const unsigned short* p = Asb + (wm * 64 + mi * 16 + l15) * TSTR;
      a[mi].u[0] = *(const v4u*)(p + h * 8);
      a[mi].u[1] = *(const v4u*)(p + 16 + h * 8);
    }
    #pragma unroll
    for (int ni = 0; ni < 2; ++ni) {
      const unsigned short* p = Bsb + (wn * 32 + ni * 16 + l15) * TSTR;
      b[ni].u[0] = *(const v4u*)(p + h * 8);
      b[ni].u[1] = *(const v4u*)(p + 16 + h * 8);
    }
    #pragma unroll
    for (int mi = 0; mi < 4; ++mi)
      #pragma unroll
      for (int ni = 0; ni < 2; ++ni)
        acc[mi][ni] = __builtin_amdgcn_wmma_f32_16x16x32_bf16(
            false, a[mi].v, false, b[ni].v, (short)0, acc[mi][ni], false, false);
  };

  if (BMODE) {
    // 4 x 16B copies per thread per tile (2 A + 2 B); wave ASYNCcnt += 4 per stage
    auto stage = [&](int buf, int kt) {
      const int k0 = kt * BK;
      unsigned short* Asb = &As[buf * BM * TSTR];
      unsigned short* Bsb = &Bs[buf * BM * TSTR];
      #pragma unroll
      for (int i = 0; i < 2; ++i) {
        const int c = tid + i * 256;                 // 0..511
        const int r = c >> 2, q16 = (c & 3) * 8;     // 4 chunks per 64B row
        cp16(Abf + (size_t)(m0 + r) * K + k0 + q16, Asb + r * TSTR + q16);
      }
      #pragma unroll
      for (int i = 0; i < 2; ++i) {
        const int c = tid + i * 256;
        const int r = c >> 2, q16 = (c & 3) * 8;
        cp16(Wt + (size_t)(n0 + r) * K + k0 + q16, Bsb + r * TSTR + q16);
      }
    };
    stage(0, 0);
    for (int kt = 0; kt < nk; ++kt) {
      const bool more = (kt + 1) < nk;
      if (more) stage((kt + 1) & 1, kt + 1);
      if (kt + 2 < nk) {                             // HBM->L2 prefetch, 2 tiles ahead
        const int kp = (kt + 2) * BK;
        __builtin_prefetch(Abf + (size_t)(m0 + (tid >> 1)) * K + kp, 0, 1);
        __builtin_prefetch(Wt  + (size_t)(n0 + (tid >> 1)) * K + kp, 0, 1);
      }
      if (more) wait_async_4(); else wait_async_0();
      __syncthreads();
      compute(&As[(kt & 1) * BM * TSTR], &Bs[(kt & 1) * BM * TSTR]);
      __syncthreads();
    }
  } else {
    for (int kt = 0; kt < nk; ++kt) {
      const int k0 = kt * BK;
      #pragma unroll
      for (int i = 0; i < 2; ++i) {                  // A: raw bf16 copy
        const int c = tid + i * 256;
        const int r = c >> 2, q16 = (c & 3) * 8;
        *(v4u*)(As + r * TSTR + q16) =
            *(const v4u*)(Abf + (size_t)(m0 + r) * K + k0 + q16);
      }
      #pragma unroll
      for (int i = 0; i < 4; ++i) {                  // B: fp32 [K,N] -> bf16 [n][k]
        const int f = tid + i * 256;
        const int r = f >> 5, c4 = (f & 31) << 2;
        const v4f bv = *(const v4f*)(Wf + (size_t)(k0 + r) * N + n0 + c4);
        Bs[(c4 + 0) * TSTR + r] = f2bf(bv[0]);
        Bs[(c4 + 1) * TSTR + r] = f2bf(bv[1]);
        Bs[(c4 + 2) * TSTR + r] = f2bf(bv[2]);
        Bs[(c4 + 3) * TSTR + r] = f2bf(bv[3]);
      }
      if (kt + 1 < nk)
        __builtin_prefetch(Wf + (size_t)(k0 + BK + (tid >> 5)) * N + n0, 0, 1);
      __syncthreads();
      compute(As, Bs);
      __syncthreads();
    }
  }

  // epilogue: C/D layout M = r + 8*h, N = l15 per 16x16 tile
  #pragma unroll
  for (int ni = 0; ni < 2; ++ni) {
    const int col = n0 + wn * 32 + ni * 16 + l15;
    const float bv = bias ? bias[col] : 0.0f;
    #pragma unroll
    for (int mi = 0; mi < 4; ++mi) {
      #pragma unroll
      for (int r = 0; r < 8; ++r) {
        const int row = m0 + wm * 64 + mi * 16 + h * 8 + r;
        float val = acc[mi][ni][r] + bv;
        if (residual) val += residual[(size_t)row * N + col];
        if (relu) val = fmaxf(val, 0.0f);
        if (Cbf) Cbf[(size_t)row * N + col] = f2bf(val);
        else     C  [(size_t)row * N + col] = val;
      }
    }
  }
}

// ---------------- weight transpose+convert: Wt[n][k] = bf16(W[k][n])
__global__ __launch_bounds__(256) void wtranspose_k(
    const float* __restrict__ W, unsigned short* __restrict__ Wt, int K, int N)
{
  __shared__ unsigned short t[32][33];
  const int k0 = blockIdx.x * 32, n0 = blockIdx.y * 32;
  const int c = threadIdx.x & 31, r0 = threadIdx.x >> 5;
  #pragma unroll
  for (int i = 0; i < 4; ++i) {
    const int r = r0 + i * 8;
    t[c][r] = f2bf(W[(size_t)(k0 + r) * N + n0 + c]);
  }
  __syncthreads();
  #pragma unroll
  for (int i = 0; i < 4; ++i) {
    const int r = r0 + i * 8;
    Wt[(size_t)(n0 + r) * K + k0 + c] = t[r][c];
  }
}

// ---------------- embedding: x[row] = tok_emb[idx[row]] + pos_emb[row % T]
__global__ __launch_bounds__(256) void embed_k(
    const int* __restrict__ idx, const float* __restrict__ tok,
    const float* __restrict__ pos, float* __restrict__ x)
{
  const int row = blockIdx.x;
  const int t = row & 1023;               // T = 1024
  const int token = idx[row];
  for (int i = threadIdx.x; i < 1024; i += 256)
    x[(size_t)row * 1024 + i] = tok[(size_t)token * 1024 + i] + pos[(size_t)t * 1024 + i];
}

// ---------------- layernorm over D per row, bf16 output (feeds GEMM A)
__global__ __launch_bounds__(256) void layernorm_k(
    const float* __restrict__ x, const float* __restrict__ s,
    const float* __restrict__ b, unsigned short* __restrict__ outbf, int D)
{
  __shared__ float red[256];
  const int row = blockIdx.x;
  const float* xr = x + (size_t)row * D;
  float sum = 0.f;
  for (int i = threadIdx.x; i < D; i += 256) sum += xr[i];
  red[threadIdx.x] = sum; __syncthreads();
  for (int st = 128; st > 0; st >>= 1) {
    if (threadIdx.x < st) red[threadIdx.x] += red[threadIdx.x + st];
    __syncthreads();
  }
  const float mean = red[0] / (float)D;
  __syncthreads();
  float vs = 0.f;
  for (int i = threadIdx.x; i < D; i += 256) { float d = xr[i] - mean; vs += d * d; }
  red[threadIdx.x] = vs; __syncthreads();
  for (int st = 128; st > 0; st >>= 1) {
    if (threadIdx.x < st) red[threadIdx.x] += red[threadIdx.x + st];
    __syncthreads();
  }
  const float rstd = rsqrtf(red[0] / (float)D + 1e-5f);
  for (int i = threadIdx.x; i < D; i += 256)
    outbf[(size_t)row * D + i] = f2bf((xr[i] - mean) * rstd * s[i] + b[i]);
}

// ---------------- causal flash attention, HD=64, one query/thread, bf16 out
// K/V chunks (32 keys) double-buffered in LDS via async global->LDS copies.
#define HD 64
#define KC 32
#define KSTR 68   // float row stride (272 B, 16B-aligned, conflict-padded)

__global__ __launch_bounds__(128) void attention_k(
    const float* __restrict__ q, const float* __restrict__ k,
    const float* __restrict__ v, unsigned short* __restrict__ o,
    int B_, int T_, int H_)
{
  __shared__ float Ks[2][KC][KSTR];
  __shared__ float Vs[2][KC][KSTR];
  const int q0 = blockIdx.x * 128;
  const int hh = blockIdx.y;
  const int bb = blockIdx.z;
  const int qi = q0 + threadIdx.x;
  const int Dm = H_ * HD;

  float qr[HD], oacc[HD];
  const float* qrow = q + ((size_t)(bb * T_ + qi)) * Dm + hh * HD;
  #pragma unroll
  for (int d = 0; d < HD; ++d) { qr[d] = qrow[d]; oacc[d] = 0.0f; }
  float mmax = -INFINITY, lsum = 0.0f;
  const float scale = 0.03125f;           // 1/sqrt(n_embed=1024) per reference

  const int nkc = (q0 + 127) >> 5;        // last needed key chunk (inclusive)

  // 8 x 16B async copies per thread per chunk (4 K + 4 V); wave ASYNCcnt += 8
  auto stage = [&](int buf, int kc) {
    const int kbase = kc * KC;
    #pragma unroll
    for (int i = 0; i < 4; ++i) {
      const int c = (int)threadIdx.x + i * 128;     // 0..511
      const int kj = c >> 4, d16 = (c & 15) * 4;    // 16 chunks per 64-float row
      const size_t off = ((size_t)(bb * T_ + kbase + kj)) * Dm + hh * HD + d16;
      cp16(k + off, &Ks[buf][kj][d16]);
      cp16(v + off, &Vs[buf][kj][d16]);
    }
  };

  stage(0, 0);
  for (int kc = 0; kc <= nkc; ++kc) {
    const bool more = (kc + 1) <= nkc;
    if (more) stage((kc + 1) & 1, kc + 1);
    if (more) wait_async_8(); else wait_async_0();
    __syncthreads();
    const int kbase = kc * KC;
    const int buf = kc & 1;
    for (int j = 0; j < KC; ++j) {
      const int kidx = kbase + j;
      if (kidx <= qi) {
        float s = 0.f;
        #pragma unroll
        for (int d = 0; d < HD; ++d) s += qr[d] * Ks[buf][j][d];
        s *= scale;
        const float mnew = fmaxf(mmax, s);
        const float corr = __expf(mmax - mnew);
        const float p    = __expf(s - mnew);
        lsum = lsum * corr + p;
        #pragma unroll
        for (int d = 0; d < HD; ++d) oacc[d] = oacc[d] * corr + p * Vs[buf][j][d];
        mmax = mnew;
      }
    }
    __syncthreads();
  }
  unsigned short* orow = o + ((size_t)(bb * T_ + qi)) * Dm + hh * HD;
  const float inv = 1.0f / lsum;
  #pragma unroll
  for (int d = 0; d < HD; ++d) orow[d] = f2bf(oacc[d] * inv);
}

// ---------------- per-row cross-entropy from logits (deterministic)
__global__ __launch_bounds__(256) void rowloss_k(
    const float* __restrict__ logits, const int* __restrict__ tgt,
    float* __restrict__ rloss, int V_)
{
  __shared__ float red[256];
  const int row = blockIdx.x;
  const float* lr = logits + (size_t)row * V_;
  float m = -INFINITY;
  for (int i = threadIdx.x; i < V_; i += 256) m = fmaxf(m, lr[i]);
  red[threadIdx.x] = m; __syncthreads();
  for (int st = 128; st > 0; st >>= 1) {
    if (threadIdx.x < st) red[threadIdx.x] = fmaxf(red[threadIdx.x], red[threadIdx.x + st]);
    __syncthreads();
  }
  m = red[0]; __syncthreads();
  float s = 0.f;
  for (int i = threadIdx.x; i < V_; i += 256) s += __expf(lr[i] - m);
  red[threadIdx.x] = s; __syncthreads();
  for (int st = 128; st > 0; st >>= 1) {
    if (threadIdx.x < st) red[threadIdx.x] += red[threadIdx.x + st];
    __syncthreads();
  }
  if (threadIdx.x == 0)
    rloss[row] = -(lr[tgt[row]] - m - __logf(red[0]));
}

__global__ __launch_bounds__(256) void lossred_k(
    const float* __restrict__ rloss, float* __restrict__ out, int n)
{
  __shared__ float red[256];
  float s = 0.f;
  for (int i = threadIdx.x; i < n; i += 256) s += rloss[i];
  red[threadIdx.x] = s; __syncthreads();
  for (int st = 128; st > 0; st >>= 1) {
    if (threadIdx.x < st) red[threadIdx.x] += red[threadIdx.x + st];
    __syncthreads();
  }
  if (threadIdx.x == 0) out[0] = red[0] / (float)n;
}

// ---------------- host orchestration ----------------
extern "C" void kernel_launch(void* const* d_in, const int* in_sizes, int n_in,
                              void* d_out, int out_size, void* d_ws, size_t ws_size,
                              hipStream_t stream)
{
  (void)in_sizes; (void)n_in; (void)out_size;

  const int*   idx     = (const int*)  d_in[0];
  const int*   targets = (const int*)  d_in[1];
  const float* tok_emb = (const float*)d_in[2];
  const float* pos_emb = (const float*)d_in[3];
  const float* Wq      = (const float*)d_in[4];
  const float* Wk      = (const float*)d_in[5];
  const float* Wv      = (const float*)d_in[6];
  const float* Wo      = (const float*)d_in[7];
  const float* bo      = (const float*)d_in[8];
  const float* ln1_s   = (const float*)d_in[9];
  const float* ln1_b   = (const float*)d_in[10];
  const float* W1      = (const float*)d_in[11];
  const float* b1      = (const float*)d_in[12];
  const float* W2      = (const float*)d_in[13];
  const float* b2      = (const float*)d_in[14];
  const float* ln2_s   = (const float*)d_in[15];
  const float* ln2_b   = (const float*)d_in[16];
  const float* lnf_s   = (const float*)d_in[17];
  const float* lnf_b   = (const float*)d_in[18];
  const float* Wlm     = (const float*)d_in[19];
  const float* blm     = (const float*)d_in[20];

  const int Dm = 1024, F4d = 4096, Tn = 1024, Bn = 4, Hn = 16, Vn = 32000, Ln = 8;
  const int BT = Bn * Tn;                  // 4096 rows
  const size_t MF = 1048576;               // 1 Mi floats

  // Scratch in the logits region of d_out (all dead before the LM-head GEMM):
  float* base = (float*)d_out;
  float* x  = base;                                        // [BT,D] fp32 @0..4M
  float* qb = base + 4 * MF;                               // fp32 @4M
  float* kb = base + 8 * MF;                               // fp32 @8M
  float* vb = base + 12 * MF;                              // fp32 @12M
  unsigned short* obf   = (unsigned short*)(base + 16 * MF);  // 4M bf16 @16M..18M
  unsigned short* ffnbf = (unsigned short*)(base + 18 * MF);  // 16M bf16 @18M..26M
  unsigned short* wT    = (unsigned short*)(base + 26 * MF);  // 12M bf16 @26M..32M
  unsigned short* wqT = wT;
  unsigned short* wkT = wT + 1 * MF;
  unsigned short* wvT = wT + 2 * MF;
  unsigned short* woT = wT + 3 * MF;
  unsigned short* w1T = wT + 4 * MF;       // [F4,D] bf16: 4M
  unsigned short* w2T = wT + 8 * MF;       // [D,F4] bf16: 4M
  float* loss = base + (size_t)BT * Vn;    // final scalar slot

  // workspace: xnbf (8 MB) + rloss (16 KB) + optional Wlm^T bf16 (64 MB)
  unsigned short* xnbf = (unsigned short*)d_ws;            // [BT,D] bf16
  const size_t xn_bytes = (size_t)BT * Dm * 2;             // 8,388,608
  float* rloss = (float*)((char*)d_ws + xn_bytes);         // [BT]
  unsigned short* wlmT =
      (unsigned short*)((char*)d_ws + xn_bytes + (size_t)BT * 4);  // 16B-aligned
  const size_t ws_need = xn_bytes + (size_t)BT * 4 + (size_t)Vn * Dm * 2;
  const bool bigws = ws_size >= ws_need;   // constant per harness -> deterministic

  const dim3 blk(256);
  const dim3 g1024(Dm  / BN, BT / BM);     // (8, 32)
  const dim3 g4096(F4d / BN, BT / BM);     // (32, 32)
  const dim3 gV   (Vn  / BN, BT / BM);     // (250, 32)

  embed_k<<<BT, 256, 0, stream>>>(idx, tok_emb, pos_emb, x);

  for (int l = 0; l < Ln; ++l) {
    const float* wq = Wq + (size_t)l * Dm * Dm;
    const float* wk = Wk + (size_t)l * Dm * Dm;
    const float* wv = Wv + (size_t)l * Dm * Dm;
    const float* wo = Wo + (size_t)l * Dm * Dm;
    const float* w1 = W1 + (size_t)l * Dm * F4d;
    const float* w2 = W2 + (size_t)l * F4d * Dm;

    // one-shot convert+transpose of this layer's weights to bf16 [N,K]
    wtranspose_k<<<dim3(Dm / 32, Dm / 32), blk, 0, stream>>>(wq, wqT, Dm, Dm);
    wtranspose_k<<<dim3(Dm / 32, Dm / 32), blk, 0, stream>>>(wk, wkT, Dm, Dm);
    wtranspose_k<<<dim3(Dm / 32, Dm / 32), blk, 0, stream>>>(wv, wvT, Dm, Dm);
    wtranspose_k<<<dim3(Dm / 32, Dm / 32), blk, 0, stream>>>(wo, woT, Dm, Dm);
    wtranspose_k<<<dim3(Dm / 32, F4d / 32), blk, 0, stream>>>(w1, w1T, Dm, F4d);
    wtranspose_k<<<dim3(F4d / 32, Dm / 32), blk, 0, stream>>>(w2, w2T, F4d, Dm);

    layernorm_k<<<BT, 256, 0, stream>>>(x, ln1_s + l * Dm, ln1_b + l * Dm, xnbf, Dm);
    gemm_wmma<1><<<g1024, blk, 0, stream>>>(xnbf, wqT, nullptr, nullptr, nullptr, qb, nullptr, BT, Dm, Dm, 0);
    gemm_wmma<1><<<g1024, blk, 0, stream>>>(xnbf, wkT, nullptr, nullptr, nullptr, kb, nullptr, BT, Dm, Dm, 0);
    gemm_wmma<1><<<g1024, blk, 0, stream>>>(xnbf, wvT, nullptr, nullptr, nullptr, vb, nullptr, BT, Dm, Dm, 0);
    attention_k<<<dim3(Tn / 128, Hn, Bn), 128, 0, stream>>>(qb, kb, vb, obf, Bn, Tn, Hn);
    // x = x + o @ Wo + bo (residual fused, in-place fp32)
    gemm_wmma<1><<<g1024, blk, 0, stream>>>(obf, woT, nullptr, bo + l * Dm, x, x, nullptr, BT, Dm, Dm, 0);
    layernorm_k<<<BT, 256, 0, stream>>>(x, ln2_s + l * Dm, ln2_b + l * Dm, xnbf, Dm);
    // ffn = relu(xn @ W1 + b1), bf16 out
    gemm_wmma<1><<<g4096, blk, 0, stream>>>(xnbf, w1T, nullptr, b1 + (size_t)l * F4d, nullptr, nullptr, ffnbf, BT, F4d, Dm, 1);
    // x = x + ffn @ W2 + b2
    gemm_wmma<1><<<g1024, blk, 0, stream>>>(ffnbf, w2T, nullptr, b2 + l * Dm, x, x, nullptr, BT, Dm, F4d, 0);
  }

  layernorm_k<<<BT, 256, 0, stream>>>(x, lnf_s, lnf_b, xnbf, Dm);
  // logits = xn @ Wlm + blm (overwrites all scratch in d_out)
  if (bigws) {
    // Wlm^T bf16 fits in d_ws (safe from logits writes) -> async bf16 path
    wtranspose_k<<<dim3(Dm / 32, Vn / 32), blk, 0, stream>>>(Wlm, wlmT, Dm, Vn);
    gemm_wmma<1><<<gV, blk, 0, stream>>>(xnbf, wlmT, nullptr, blm, nullptr, base, nullptr, BT, Vn, Dm, 0);
  } else {
    gemm_wmma<0><<<gV, blk, 0, stream>>>(xnbf, nullptr, Wlm, blm, nullptr, base, nullptr, BT, Vn, Dm, 0);
  }
  rowloss_k<<<BT, 256, 0, stream>>>(base, targets, rloss, Vn);
  lossred_k<<<1, 256, 0, stream>>>(rloss, loss, BT);
}
